// JordanNetworkScratch_44908178047580
// MI455X (gfx1250) — matrix operand order repaired and checked by reading, not compiled, not run on hardware
//
#include <hip/hip_runtime.h>
#include <hip/hip_bf16.h>
#include <math.h>

// ---------------- problem constants ----------------
#define IN_SZ   2048
#define HID_SZ  4096
#define OUT_SZ  2048
#define T_SZ    2048
#define NBLK    128          // persistent blocks for the recurrent kernel
#define BTHR    256          // 8 wave32 per block

typedef __bf16 v8bf  __attribute__((ext_vector_type(8)));
typedef __bf16 v16bf __attribute__((ext_vector_type(16)));
typedef float  v8f   __attribute__((ext_vector_type(8)));

// ---------------- fp32 -> bf16 convert (vector x4) ----------------
__global__ void cvt_bf16_kernel(const float* __restrict__ src,
                                __bf16* __restrict__ dst, int n) {
    int i = (blockIdx.x * blockDim.x + threadIdx.x) * 4;
    if (i < n) {
        float4 f = *(const float4*)(src + i);
        dst[i + 0] = (__bf16)f.x;
        dst[i + 1] = (__bf16)f.y;
        dst[i + 2] = (__bf16)f.z;
        dst[i + 3] = (__bf16)f.w;
    }
}

// ---------------- barrier-state init (runs every launch -> deterministic) ----
__global__ void init_bar_kernel(unsigned* bar) {
    bar[0] = 0u;   // arrival counter
    bar[1] = 0u;   // generation
}

// ---------------- Phase A: P[t][j] = sum_k x[t][k]*Wx[j][k] + bh[j] --------
// One wave per 16x16 output tile; K-loop in steps of 32 with
// v_wmma_f32_16x16x32_bf16. Fragments are loaded straight from global:
//   A (16x32, bf16):  lane m=lane&15, hi=lane>>4
//       elems 0..7  = X[t0+m][k0 + hi*8 + e]
//       elems 8..15 = X[t0+m][k0 + 16 + hi*8 + e]
//   B (32x16, bf16):  lane n=lane&15, hi=lane>>4
//       elems 0..7  = Wx[j0+n][k0 + hi*8 + e]   (row-major Wx == B^T tile)
//       elems 8..15 = Wx[j0+n][k0 + 16 + hi*8 + e]
__global__ void precompute_gemm_kernel(const __bf16* __restrict__ Xb,   // [T][IN]
                                       const __bf16* __restrict__ Wxb,  // [HID][IN]
                                       const float*  __restrict__ bh,   // [HID]
                                       float* __restrict__ P) {         // [T][HID]
    const int lane = threadIdx.x & 31;
    const int wave = blockIdx.x * (BTHR >> 5) + (threadIdx.x >> 5);
    const int jt = wave & 255;          // 256 column tiles (HID/16)
    const int tt = wave >> 8;           // 128 row tiles    (T/16)
    const int m  = lane & 15;
    const int hi = lane >> 4;

    const __bf16* arow = Xb  + (size_t)(tt * 16 + m) * IN_SZ + hi * 8;
    const __bf16* brow = Wxb + (size_t)(jt * 16 + m) * IN_SZ + hi * 8;

    v8f c = {};
#pragma unroll 4
    for (int k0 = 0; k0 < IN_SZ; k0 += 32) {
        // locality 3 -> WGP-scope prefetch (pull into all cache levels)
        __builtin_prefetch(arow + k0 + 256, 0, 3);
        __builtin_prefetch(brow + k0 + 256, 0, 3);
        v8bf a_lo = *(const v8bf*)(arow + k0);
        v8bf a_hi = *(const v8bf*)(arow + k0 + 16);
        v8bf b_lo = *(const v8bf*)(brow + k0);
        v8bf b_hi = *(const v8bf*)(brow + k0 + 16);
        v16bf a, b;
#pragma unroll
        for (int e = 0; e < 8; ++e) {
            a[e] = a_lo[e]; a[e + 8] = a_hi[e];
            b[e] = b_lo[e]; b[e + 8] = b_hi[e];
        }
        c = __builtin_amdgcn_wmma_f32_16x16x32_bf16(
                false, a, false, b, (short)0, c, false, false);
    }

    // C/D layout: VGPR r, lane l -> M = r + (l>>4)*8, N = l&15
    const int   n    = lane & 15;
    const float bias = bh[jt * 16 + n];
    float* prow = P + (size_t)(tt * 16) * HID_SZ + jt * 16 + n;
#pragma unroll
    for (int r = 0; r < 8; ++r)
        prow[(size_t)(r + hi * 8) * HID_SZ] = c[r] + bias;
}

// ---------------- async global->LDS staging ------------------------------
// `smem` is the only __shared__ object in the kernel, so its LDS byte offset
// is 0: we can pass raw byte offsets as both the LDS address (VDST) and the
// VGPR part of the GVS global address. Completion via ASYNCcnt.
__device__ __forceinline__ void async_stage(const float* __restrict__ gsrc,
                                            int bytes, int tid) {
    for (int off = tid * 16; off < bytes; off += BTHR * 16) {
        asm volatile("global_load_async_to_lds_b128 %0, %1, %2"
                     :: "v"(off), "v"(off), "s"(gsrc) : "memory");
    }
    asm volatile("s_wait_asynccnt 0" ::: "memory");
}

// ---------------- grid-wide barrier (persistent kernel) -------------------
__device__ __forceinline__ void grid_barrier(unsigned* cnt, unsigned* gen,
                                             unsigned nblocks) {
    __threadfence();           // make this thread's global writes visible
    __syncthreads();
    if (threadIdx.x == 0) {
        unsigned g = __hip_atomic_load(gen, __ATOMIC_ACQUIRE,
                                       __HIP_MEMORY_SCOPE_AGENT);
        unsigned a = __hip_atomic_fetch_add(cnt, 1u, __ATOMIC_ACQ_REL,
                                            __HIP_MEMORY_SCOPE_AGENT);
        if (a == nblocks - 1u) {
            __hip_atomic_store(cnt, 0u, __ATOMIC_RELAXED,
                               __HIP_MEMORY_SCOPE_AGENT);
            __hip_atomic_fetch_add(gen, 1u, __ATOMIC_ACQ_REL,
                                   __HIP_MEMORY_SCOPE_AGENT);
        } else {
            while (__hip_atomic_load(gen, __ATOMIC_ACQUIRE,
                                     __HIP_MEMORY_SCOPE_AGENT) == g)
                __builtin_amdgcn_s_sleep(1);
        }
    }
    __syncthreads();
}

// ---------------- Phase B: the sequential recurrence ----------------------
// h_t = tanh(P[t] + Wh @ y_{t-1});  y_t = Wy @ h_t + by
// One wave per PAIR of output rows: two independent 16B/lane bf16 weight
// streams in flight (hides L2 latency), activation LDS reads amortized over
// both rows. __shfl_xor wave reduction. Weights stream from L2 (bf16 copies
// are 32 MB total, resident in the 192 MB L2).
__global__ void recurrent_kernel(const float*  __restrict__ P,    // [T][HID] (incl. bh)
                                 const __bf16* __restrict__ Whb,  // [HID][OUT]
                                 const __bf16* __restrict__ Wyb,  // [OUT][HID]
                                 const float*  __restrict__ by,   // [OUT]
                                 float* __restrict__ hbuf,        // [HID] scratch
                                 float* __restrict__ y,           // [T][OUT] = d_out
                                 unsigned* __restrict__ bar) {
    __shared__ float smem[HID_SZ];               // 16 KB, base offset 0 in LDS
    const int tid  = threadIdx.x;
    const int lane = tid & 31;
    const int wid  = blockIdx.x * (BTHR >> 5) + (tid >> 5);
    const int nw   = NBLK * (BTHR >> 5);         // 1024 waves

    for (int t = 0; t < T_SZ; ++t) {
        // ---- stage y_{t-1} into LDS (8 KB) ----
        if (t == 0) {
            for (int i = tid; i < OUT_SZ; i += BTHR) smem[i] = 0.0f;
        } else {
            async_stage(y + (size_t)(t - 1) * OUT_SZ, OUT_SZ * 4, tid);
        }
        __syncthreads();

        // ---- h = tanh(P[t] + Wh @ y_prev) : 4096 rows, 2 per wave-pass ----
        for (int row = wid * 2; row < HID_SZ; row += nw * 2) {
            const __bf16* w0 = Whb + (size_t)row * OUT_SZ;
            const __bf16* w1 = w0 + OUT_SZ;
            float acc0 = 0.0f, acc1 = 0.0f;
#pragma unroll
            for (int k = lane * 8; k < OUT_SZ; k += 32 * 8) {
                v8bf a8 = *(const v8bf*)(w0 + k);
                v8bf b8 = *(const v8bf*)(w1 + k);
#pragma unroll
                for (int e = 0; e < 8; ++e) {
                    float s = smem[k + e];
                    acc0 += (float)a8[e] * s;
                    acc1 += (float)b8[e] * s;
                }
            }
#pragma unroll
            for (int off = 16; off; off >>= 1) {
                acc0 += __shfl_xor(acc0, off, 32);
                acc1 += __shfl_xor(acc1, off, 32);
            }
            if (lane == 0) {
                const float* prow = P + (size_t)t * HID_SZ + row;
                hbuf[row + 0] = tanhf(prow[0] + acc0);
                hbuf[row + 1] = tanhf(prow[1] + acc1);
            }
        }
        grid_barrier(bar, bar + 1, NBLK);

        // ---- stage h into LDS (16 KB) ----
        async_stage(hbuf, HID_SZ * 4, tid);
        __syncthreads();

        // ---- y_t = Wy @ h + by : 2048 rows, 2 per wave-pass ----
        for (int row = wid * 2; row < OUT_SZ; row += nw * 2) {
            const __bf16* w0 = Wyb + (size_t)row * HID_SZ;
            const __bf16* w1 = w0 + HID_SZ;
            float acc0 = 0.0f, acc1 = 0.0f;
#pragma unroll
            for (int k = lane * 8; k < HID_SZ; k += 32 * 8) {
                v8bf a8 = *(const v8bf*)(w0 + k);
                v8bf b8 = *(const v8bf*)(w1 + k);
#pragma unroll
                for (int e = 0; e < 8; ++e) {
                    float s = smem[k + e];
                    acc0 += (float)a8[e] * s;
                    acc1 += (float)b8[e] * s;
                }
            }
#pragma unroll
            for (int off = 16; off; off >>= 1) {
                acc0 += __shfl_xor(acc0, off, 32);
                acc1 += __shfl_xor(acc1, off, 32);
            }
            if (lane == 0) {
                float* yrow = y + (size_t)t * OUT_SZ + row;
                yrow[0] = by[row + 0] + acc0;
                yrow[1] = by[row + 1] + acc1;
            }
        }
        grid_barrier(bar, bar + 1, NBLK);
    }
}

// ---------------- launcher ------------------------------------------------
extern "C" void kernel_launch(void* const* d_in, const int* in_sizes, int n_in,
                              void* d_out, int out_size, void* d_ws, size_t ws_size,
                              hipStream_t stream) {
    const float* x  = (const float*)d_in[0];   // [T][IN]
    const float* Wx = (const float*)d_in[1];   // [HID][IN]
    const float* Wh = (const float*)d_in[2];   // [HID][OUT]
    const float* Wy = (const float*)d_in[3];   // [OUT][HID]
    const float* bh = (const float*)d_in[4];   // [HID]
    const float* by = (const float*)d_in[5];   // [OUT]
    float* y = (float*)d_out;                  // [T][OUT] fp32

    // workspace layout (all 16B aligned; ~92 MB total)
    char* ws = (char*)d_ws;
    unsigned* bar = (unsigned*)ws;                              // 256 B
    __bf16* Xb  = (__bf16*)(ws + 256);                          // 8 MB
    __bf16* Wxb = Xb  + (size_t)T_SZ  * IN_SZ;                  // 16 MB
    __bf16* Whb = Wxb + (size_t)HID_SZ * IN_SZ;                 // 16 MB
    __bf16* Wyb = Whb + (size_t)HID_SZ * OUT_SZ;                // 16 MB
    float*  P   = (float*)(Wyb + (size_t)OUT_SZ * HID_SZ);      // 32 MB
    float*  hbuf = P + (size_t)T_SZ * HID_SZ;                   // 16 KB

    init_bar_kernel<<<1, 1, 0, stream>>>(bar);

    // fp32 -> bf16 (4 elements per thread)
    int nX  = T_SZ * IN_SZ, nWx = HID_SZ * IN_SZ,
        nWh = HID_SZ * OUT_SZ, nWy = OUT_SZ * HID_SZ;
    cvt_bf16_kernel<<<nX  / 1024, 256, 0, stream>>>(x,  Xb,  nX);
    cvt_bf16_kernel<<<nWx / 1024, 256, 0, stream>>>(Wx, Wxb, nWx);
    cvt_bf16_kernel<<<nWh / 1024, 256, 0, stream>>>(Wh, Whb, nWh);
    cvt_bf16_kernel<<<nWy / 1024, 256, 0, stream>>>(Wy, Wyb, nWy);

    // Phase A: 32768 tiles / 8 waves per block = 4096 blocks
    precompute_gemm_kernel<<<4096, BTHR, 0, stream>>>(Xb, Wxb, bh, P);

    // Phase B: persistent recurrence
    recurrent_kernel<<<NBLK, BTHR, 0, stream>>>(P, Whb, Wyb, by, hbuf, y, bar);
}
